// SGFormer_1949915152402
// MI455X (gfx1250) — matrix operand order repaired and checked by reading, not compiled, not run on hardware
//
#include <hip/hip_runtime.h>

// ---------------------------------------------------------------------------
// SGFormer forward for MI455X (gfx1250, wave32, WMMA bf16->f32 accumulate).
//
// Input flattening assumption (setup_inputs insertion order, params nested
// insertion order):
//  0 x[65536*128] f32, 1 edge_index[2*1048576] i32, 2 batch[65536] i32,
//  3 g_fc_w, 4 g_fc_b, 5 g_bn0_g, 6 g_bn0_b, 7 t_fc_w, 8 t_fc_b,
//  9 t_ln0_g, 10 t_ln0_b, 11 fc_w[128*64], 12 fc_b[64],
//  13.. g_blocks x3 {w,b,bn_g,bn_b}, 25.. t_blocks x2 {q,k,v,ln_g,ln_b}
//
// Workspace: ~228 MB (4 MB small region + 7 x 32 MB activation buffers).
// ---------------------------------------------------------------------------

#define NNODES 65536
#define NEDGES 1048576
#define NGRAPH 8
#define NPER   8192
#define HID    128
#define OUTC   64
#define EPSL   1e-5f

typedef __attribute__((ext_vector_type(16))) __bf16 v16bf;
typedef __attribute__((ext_vector_type(4)))  __bf16 bf4;
typedef __attribute__((ext_vector_type(8)))  float  v8f;

__device__ __forceinline__ __bf16 f2bf(float f) { return (__bf16)f; }

__device__ __forceinline__ float wave_sum(float x) {
  #pragma unroll
  for (int o = 16; o > 0; o >>= 1) x += __shfl_xor(x, o, 32);
  return x;
}
__device__ __forceinline__ float wave_max(float x) {
  #pragma unroll
  for (int o = 16; o > 0; o >>= 1) x = fmaxf(x, __shfl_xor(x, o, 32));
  return x;
}

// Unused A-tile LDS permutation (A tiles use natural k): per 32-k block, order [k0..7,k16..23,k8..15,k24..31]
// so each (half,ks) fragment is 16 contiguous bf16. submap for 8-k subgroup id.
__device__ __forceinline__ int a_sub(int sub) {   // sub = (k>>3)&3 -> 0,16,8,24
  return ((sub & 1) << 4) | ((sub & 2) << 2);
}

// ---------------------------------------------------------------------------
// WMMA GEMM: C[N x COUT] = A[N x 128] @ W[128 x COUT]  (bf16 in, f32 acc)
// Block = 128 rows x COUT cols, 256 threads = 8 waves, wave -> 16-row strip.
// MODE 0: none; 1: +b1; 2: relu((y+b1)*(sc*rsq)+b2)  (folded BN-eval).
// PERGRAPH: W points at 8 stacked 128x128 matrices, selected by row block.
// ---------------------------------------------------------------------------
template<int NT, int MODE, bool PERGRAPH>
__global__ __launch_bounds__(256) void
gemm_k128(const float* __restrict__ A, const float* __restrict__ W,
          float* __restrict__ C, const float* __restrict__ b1,
          const float* __restrict__ sc, const float* __restrict__ b2)
{
  constexpr int COUT = NT * 16;
  constexpr int LOG2C = (COUT == 128) ? 7 : 6;
  constexpr int APITCH = 144;                       // 288 B rows, 32B aligned
  __shared__ __align__(32) __bf16 lds_a[128 * APITCH];
  __shared__ __align__(32) __bf16 lds_w[128 * COUT];  // fragment-major
  const int tid = threadIdx.x;
  const long row0 = (long)blockIdx.x * 128;
  const float* Wp = PERGRAPH ? (W + (size_t)(blockIdx.x >> 6) * 128 * 128) : W;

  // stage A tile 128x128: float4 loads, bf4 stores, fragment-major k order
  #pragma unroll
  for (int i = 0; i < 16; i++) {
    int lin = i * 256 + tid;
    int row = lin >> 5, c4 = (lin & 31) * 4;
    float4 f = *(const float4*)&A[(row0 + row) * 128 + c4];
    int dst = row * APITCH + c4;
    bf4 p = { f2bf(f.x), f2bf(f.y), f2bf(f.z), f2bf(f.w) };
    *(bf4*)&lds_a[dst] = p;
  }
  // stage W (128 x COUT): 4 strided k-reads per column, fragment-major
  #pragma unroll
  for (int i = 0; i < COUT / 8; i++) {
    int chunk = i * 256 + tid;
    int c = chunk & (COUT - 1), k0 = (chunk >> LOG2C) * 4;
    bf4 p = { f2bf(Wp[(k0 + 0) * COUT + c]), f2bf(Wp[(k0 + 1) * COUT + c]),
              f2bf(Wp[(k0 + 2) * COUT + c]), f2bf(Wp[(k0 + 3) * COUT + c]) };
    *(bf4*)&lds_w[((k0 >> 4) * COUT + c) * 16 + (k0 & 15)] = p;
  }
  __syncthreads();

  const int wave = tid >> 5, lane = tid & 31;
  const int m = lane & 15, half = lane >> 4, m0 = wave * 16;
  v8f acc[NT];
  #pragma unroll
  for (int ct = 0; ct < NT; ct++)
    #pragma unroll
    for (int j = 0; j < 8; j++) acc[ct][j] = 0.0f;

  #pragma unroll
  for (int ks = 0; ks < 4; ks++) {
    v16bf af = *(const v16bf*)&lds_a[(m0 + m) * APITCH + ks * 32 + half * 16];
    #pragma unroll
    for (int ct = 0; ct < NT; ct++) {
      v16bf bfv = *(const v16bf*)&lds_w[((ks * 2 + half) * COUT + ct * 16 + m) * 16];
      acc[ct] = __builtin_amdgcn_wmma_f32_16x16x32_bf16(
          false, af, false, bfv, (short)0, acc[ct], false, false);
    }
  }
  const float rsq = 0.9999950000374997f;            // rsqrt(1+1e-5)
  #pragma unroll
  for (int ct = 0; ct < NT; ct++)
    #pragma unroll
    for (int r = 0; r < 8; r++) {
      int rowl = m0 + half * 8 + r, col = ct * 16 + m;
      float val = acc[ct][r];
      if (MODE == 1) val += b1[col];
      if (MODE == 2) val = fmaxf(0.f, (val + b1[col]) * (sc[col] * rsq) + b2[col]);
      C[(row0 + rowl) * COUT + col] = val;
    }
}

// kvs[g] = K_g^T @ V_g  (per-graph 128x128, K-dim 8192, split-K=8, atomics)
__global__ __launch_bounds__(256) void
gemm_tn_kvs(const float* __restrict__ K, const float* __restrict__ V,
            float* __restrict__ kvs)
{
  constexpr int KTP = 48;                            // 96 B pitch, 32B aligned
  __shared__ __align__(32) __bf16 ldsKt[128 * KTP];  // K^T chunk, frag-major
  __shared__ __align__(32) __bf16 ldsVp[2 * 128 * 16];
  const int g = blockIdx.x >> 3, s = blockIdx.x & 7;
  const long r0 = (long)g * NPER + (long)s * 1024;
  const int tid = threadIdx.x, wave = tid >> 5, lane = tid & 31;
  const int m = lane & 15, half = lane >> 4, m0 = wave * 16;
  v8f acc[8];
  #pragma unroll
  for (int ct = 0; ct < 8; ct++)
    #pragma unroll
    for (int j = 0; j < 8; j++) acc[ct][j] = 0.0f;

  for (int step = 0; step < 32; step++) {
    __syncthreads();
    long rb = r0 + step * 32;
    // 32 rows x 128 cols; thread reads 4 k's down one column of K and V
    #pragma unroll
    for (int i = 0; i < 4; i++) {
      int chunk = i * 256 + tid;
      int c = chunk & 127, r4 = (chunk >> 7) * 4;
      bf4 pk = { f2bf(K[(rb + r4 + 0) * 128 + c]), f2bf(K[(rb + r4 + 1) * 128 + c]),
                 f2bf(K[(rb + r4 + 2) * 128 + c]), f2bf(K[(rb + r4 + 3) * 128 + c]) };
      int l0 = r4;
      *(bf4*)&ldsKt[c * KTP + l0] = pk;              // A-fragment order over k=r
      bf4 pv = { f2bf(V[(rb + r4 + 0) * 128 + c]), f2bf(V[(rb + r4 + 1) * 128 + c]),
                 f2bf(V[(rb + r4 + 2) * 128 + c]), f2bf(V[(rb + r4 + 3) * 128 + c]) };
      *(bf4*)&ldsVp[((r4 >> 4) * 128 + c) * 16 + (r4 & 15)] = pv;
    }
    __syncthreads();
    v16bf af = *(const v16bf*)&ldsKt[(m0 + m) * KTP + half * 16];
    #pragma unroll
    for (int ct = 0; ct < 8; ct++) {
      v16bf bfv = *(const v16bf*)&ldsVp[(half * 128 + ct * 16 + m) * 16];
      acc[ct] = __builtin_amdgcn_wmma_f32_16x16x32_bf16(
          false, af, false, bfv, (short)0, acc[ct], false, false);
    }
  }
  #pragma unroll
  for (int ct = 0; ct < 8; ct++)
    #pragma unroll
    for (int r = 0; r < 8; r++) {
      int rowl = m0 + half * 8 + r, col = ct * 16 + m;
      atomicAdd(&kvs[(size_t)g * 16384 + rowl * 128 + col], acc[ct][r]);
    }
}

// ----------------------------- utility kernels -----------------------------
__global__ void zero_f32(float* __restrict__ p, long n) {
  long i = (long)blockIdx.x * blockDim.x + threadIdx.x;
  if (i < n) p[i] = 0.0f;
}
__global__ void deg_init(float* __restrict__ deg) {
  int i = blockIdx.x * blockDim.x + threadIdx.x;
  deg[i] = 1.0f;                                  // self-loop
}
__global__ void deg_edge(const int* __restrict__ ei, float* __restrict__ deg) {
  long e = (long)blockIdx.x * blockDim.x + threadIdx.x;
  atomicAdd(&deg[ei[NEDGES + e]], 1.0f);          // col = dest
}
__global__ void dis_kernel(const float* __restrict__ deg, float* __restrict__ dis) {
  int i = blockIdx.x * blockDim.x + threadIdx.x;
  dis[i] = rsqrtf(deg[i]);                        // deg >= 1 always
}

// --------- stable counting sort by batch id (3 passes, NB=256) -------------
__global__ void hist_kernel(const int* __restrict__ batch, int* __restrict__ counts) {
  __shared__ int cnt[NGRAPH];
  int tid = threadIdx.x;
  if (tid < NGRAPH) cnt[tid] = 0;
  __syncthreads();
  atomicAdd(&cnt[batch[blockIdx.x * 256 + tid]], 1);
  __syncthreads();
  if (tid < NGRAPH) counts[tid * 256 + blockIdx.x] = cnt[tid];
}
__global__ void scan_kernel(const int* __restrict__ counts, int* __restrict__ offs) {
  if (threadIdx.x == 0) {
    int run = 0;
    for (int i = 0; i < NGRAPH * 256; i++) { offs[i] = run; run += counts[i]; }
  }
}
__global__ void scatter_kernel(const int* __restrict__ batch, const int* __restrict__ offs,
                               int* __restrict__ idx, int* __restrict__ rev) {
  if (threadIdx.x != 0) return;
  int blk = blockIdx.x, local[NGRAPH];
  for (int b = 0; b < NGRAPH; b++) local[b] = offs[b * 256 + blk];
  int base = blk * 256;
  for (int i = 0; i < 256; i++) {
    int node = base + i, b = batch[node];
    int pos = local[b]++;
    idx[pos] = node; rev[node] = pos;
  }
}
__global__ void gather_rows(const float* __restrict__ x, const int* __restrict__ idx,
                            float* __restrict__ xs) {
  long gid = (long)blockIdx.x * blockDim.x + threadIdx.x;   // N*32 threads
  long p = gid >> 5; int c4 = (int)(gid & 31);
  ((float4*)xs)[p * 32 + c4] = ((const float4*)x)[(long)idx[p] * 32 + c4];
}

// ------------------------------- GCN branch --------------------------------
__global__ void gcn_init_agg(const float* __restrict__ hw, const float* __restrict__ dis,
                             float* __restrict__ agg) {
  long gid = (long)blockIdx.x * blockDim.x + threadIdx.x;
  long node = gid >> 5; int c4 = (int)(gid & 31);
  float d2 = dis[node]; d2 *= d2;
  float4 v = ((const float4*)hw)[node * 32 + c4];
  v.x *= d2; v.y *= d2; v.z *= d2; v.w *= d2;
  ((float4*)agg)[node * 32 + c4] = v;
}
__global__ __launch_bounds__(256) void
gcn_edge(const int* __restrict__ ei, const float* __restrict__ dis,
         const float* __restrict__ hw, float* __restrict__ agg) {
  int wave = threadIdx.x >> 5, lane = threadIdx.x & 31;
  long e = (long)blockIdx.x * 8 + wave;
  int r = ei[e], c = ei[NEDGES + e];
  float nrm = dis[r] * dis[c];
  #pragma unroll
  for (int j = 0; j < 4; j++) {
    int ch = lane + 32 * j;
    atomicAdd(&agg[(long)c * 128 + ch], hw[(long)r * 128 + ch] * nrm);
  }
}
__global__ void gcn_post(const float* __restrict__ agg, const float* __restrict__ b,
                         const float* __restrict__ bng, const float* __restrict__ bnb,
                         float* __restrict__ h) {
  const float rsq = 0.9999950000374997f;
  long gid = (long)blockIdx.x * blockDim.x + threadIdx.x;
  long node = gid >> 5; int c4 = (int)(gid & 31);
  float4 a = ((const float4*)agg)[node * 32 + c4];
  float4 ho = ((float4*)h)[node * 32 + c4];
  float* ap = (float*)&a; float* hp = (float*)&ho;
  #pragma unroll
  for (int l = 0; l < 4; l++) {
    int c = c4 * 4 + l;
    float v = fmaxf(0.f, (ap[l] + b[c]) * (bng[c] * rsq) + bnb[c]);
    hp[l] = v + hp[l];                           // residual; h becomes "last"
  }
  ((float4*)h)[node * 32 + c4] = ho;
}

// --------------------------- transformer branch ----------------------------
__global__ __launch_bounds__(256) void
ln_relu_kernel(float* __restrict__ t, const float* __restrict__ g,
               const float* __restrict__ b) {
  int wave = threadIdx.x >> 5, lane = threadIdx.x & 31;
  long node = (long)blockIdx.x * 8 + wave;
  float y[4], s = 0.f;
  #pragma unroll
  for (int j = 0; j < 4; j++) { y[j] = t[node * 128 + lane + 32 * j]; s += y[j]; }
  float mean = wave_sum(s) * (1.f / 128.f);
  float vs = 0.f;
  #pragma unroll
  for (int j = 0; j < 4; j++) { float d = y[j] - mean; vs += d * d; }
  float rstd = rsqrtf(wave_sum(vs) * (1.f / 128.f) + EPSL);
  #pragma unroll
  for (int j = 0; j < 4; j++) {
    int c = lane + 32 * j;
    t[node * 128 + c] = fmaxf(0.f, (y[j] - mean) * rstd * g[c] + b[c]);
  }
}
__global__ void sumsq_kernel(const float* __restrict__ X, float* __restrict__ acc) {
  long i = (long)blockIdx.x * blockDim.x + threadIdx.x;
  float s = 0.f;
  const long n4 = (long)NNODES * 32;
  for (long p = i; p < n4; p += (long)gridDim.x * blockDim.x) {
    float4 f = ((const float4*)X)[p];
    s += f.x * f.x + f.y * f.y + f.z * f.z + f.w * f.w;
  }
  s = wave_sum(s);
  if ((threadIdx.x & 31) == 0) atomicAdd(acc, s);
}
__global__ void colsum_kernel(const float* __restrict__ K, float* __restrict__ ks) {
  int g = blockIdx.x >> 3, ch = blockIdx.x & 7, c = threadIdx.x;   // 128 thr
  long base = ((long)g * NPER + ch * 1024) * 128;
  float s = 0.f;
  for (int r = 0; r < 1024; r++) s += K[base + (long)r * 128 + c];
  atomicAdd(&ks[g * 128 + c], s);
}
__global__ __launch_bounds__(256) void
attn_combine(const float* __restrict__ P, const float* __restrict__ q,
             const float* __restrict__ v, float* __restrict__ t,
             const float* __restrict__ ks, const float* __restrict__ scal,
             const float* __restrict__ lng, const float* __restrict__ lnb) {
  int wave = threadIdx.x >> 5, lane = threadIdx.x & 31;
  long node = (long)blockIdx.x * 8 + wave;
  int g = (int)(node >> 13);
  float inv = rsqrtf(scal[0]) * rsqrtf(scal[1]);    // 1/(||q|| * ||k||)
  float qv[4], dp = 0.f;
  #pragma unroll
  for (int j = 0; j < 4; j++) {
    int c = lane + 32 * j;
    qv[j] = q[node * 128 + c];
    dp += qv[j] * ks[g * 128 + c];
  }
  float den = wave_sum(dp) * inv + (float)NPER;
  float y[4], msum = 0.f;
  #pragma unroll
  for (int j = 0; j < 4; j++) {
    int c = lane + 32 * j;
    float num = P[node * 128 + c] * inv + (float)NPER * v[node * 128 + c];
    float a = num / den;
    a = (a + t[node * 128 + c]) * 0.5f;
    y[j] = a; msum += a;
  }
  float mean = wave_sum(msum) * (1.f / 128.f);
  float vs = 0.f;
  #pragma unroll
  for (int j = 0; j < 4; j++) { float d = y[j] - mean; vs += d * d; }
  float rstd = rsqrtf(wave_sum(vs) * (1.f / 128.f) + EPSL);
  #pragma unroll
  for (int j = 0; j < 4; j++) {
    int c = lane + 32 * j;
    t[node * 128 + c] = fmaxf(0.f, (y[j] - mean) * rstd * lng[c] + lnb[c]);
  }
}

// ------------------------------- combine/out -------------------------------
__global__ void combine_out(const float* __restrict__ h, const float* __restrict__ ts,
                            const int* __restrict__ rev, float* __restrict__ out) {
  long gid = (long)blockIdx.x * blockDim.x + threadIdx.x;
  long node = gid >> 5; int c4 = (int)(gid & 31);
  float4 a = ((const float4*)h)[node * 32 + c4];
  float4 b = ((const float4*)ts)[(long)rev[node] * 32 + c4];
  a.x = 0.5f * (a.x + b.x); a.y = 0.5f * (a.y + b.y);
  a.z = 0.5f * (a.z + b.z); a.w = 0.5f * (a.w + b.w);
  ((float4*)out)[node * 32 + c4] = a;
}
__global__ __launch_bounds__(256) void
log_softmax64(const float* __restrict__ logits, float* __restrict__ out) {
  int wave = threadIdx.x >> 5, lane = threadIdx.x & 31;
  long node = (long)blockIdx.x * 8 + wave;
  float a = logits[node * 64 + lane], b = logits[node * 64 + 32 + lane];
  float mx = wave_max(fmaxf(a, b));
  float s = wave_sum(expf(a - mx) + expf(b - mx));
  float l = logf(s);
  out[node * 64 + lane]      = a - mx - l;
  out[node * 64 + 32 + lane] = b - mx - l;
}

// ---------------------------------------------------------------------------
extern "C" void kernel_launch(void* const* d_in, const int* in_sizes, int n_in,
                              void* d_out, int out_size, void* d_ws, size_t ws_size,
                              hipStream_t stream) {
  (void)in_sizes; (void)n_in; (void)out_size; (void)ws_size;
  const float* x      = (const float*)d_in[0];
  const int*   ei     = (const int*)d_in[1];
  const int*   batch  = (const int*)d_in[2];
  const float* g_fc_w = (const float*)d_in[3];
  const float* g_fc_b = (const float*)d_in[4];
  const float* g_bn0g = (const float*)d_in[5];
  const float* g_bn0b = (const float*)d_in[6];
  const float* t_fc_w = (const float*)d_in[7];
  const float* t_fc_b = (const float*)d_in[8];
  const float* t_ln0g = (const float*)d_in[9];
  const float* t_ln0b = (const float*)d_in[10];
  const float* fc_w   = (const float*)d_in[11];
  const float* fc_b   = (const float*)d_in[12];
  const float *gw[3], *gb[3], *gbng[3], *gbnb[3];
  for (int l = 0; l < 3; l++) {
    gw[l]   = (const float*)d_in[13 + l * 4 + 0];
    gb[l]   = (const float*)d_in[13 + l * 4 + 1];
    gbng[l] = (const float*)d_in[13 + l * 4 + 2];
    gbnb[l] = (const float*)d_in[13 + l * 4 + 3];
  }
  const float *tq[2], *tk[2], *tv[2], *tlng[2], *tlnb[2];
  for (int l = 0; l < 2; l++) {
    tq[l]   = (const float*)d_in[25 + l * 5 + 0];
    tk[l]   = (const float*)d_in[25 + l * 5 + 1];
    tv[l]   = (const float*)d_in[25 + l * 5 + 2];
    tlng[l] = (const float*)d_in[25 + l * 5 + 3];
    tlnb[l] = (const float*)d_in[25 + l * 5 + 4];
  }

  // workspace layout (floats)
  float* deg   = (float*)d_ws;
  float* dis   = deg + NNODES;
  int*   idxb  = (int*)(dis + NNODES);
  int*   revb  = idxb + NNODES;
  int*   cnts  = revb + NNODES;                  // 2048
  int*   offs  = cnts + 2048;                    // 2048
  float* kvs   = (float*)(offs + 2048);          // 8*128*128
  float* ksum  = kvs + NGRAPH * 128 * 128;       // 8*128
  float* scal  = ksum + NGRAPH * 128;            // 2 scalars
  float* big   = (float*)d_ws + (1 << 20);       // 4 MB offset
  const size_t BB = (size_t)NNODES * HID;        // 8388608 floats
  float* h   = big + 0 * BB;
  float* hw  = big + 1 * BB;
  float* agg = big + 2 * BB;
  float* ts  = big + 3 * BB;
  float* qb  = big + 4 * BB;    // also xs (free before q is produced)
  float* kb  = big + 5 * BB;    // also P  (k dead once kvs/ks/||k|| done)
  float* vb  = big + 6 * BB;
  float* xs = qb, *P = kb, *outc = agg, *logits = hw;

  const dim3 B256(256);
  const int GB_N32 = (NNODES * 32) / 256;        // float4 elementwise grid
  const int GB_GEMM = NNODES / 128;              // 512

  // ---- degrees / symmetric norm ----
  deg_init<<<NNODES / 256, B256, 0, stream>>>(deg);
  deg_edge<<<NEDGES / 256, B256, 0, stream>>>(ei, deg);
  dis_kernel<<<NNODES / 256, B256, 0, stream>>>(deg, dis);

  // ---- stable counting sort by batch (for contiguous per-graph GEMMs) ----
  hist_kernel<<<256, B256, 0, stream>>>(batch, cnts);
  scan_kernel<<<1, 32, 0, stream>>>(cnts, offs);
  scatter_kernel<<<256, 32, 0, stream>>>(batch, offs, idxb, revb);
  gather_rows<<<GB_N32, B256, 0, stream>>>(x, idxb, xs);

  // ---- GNN branch: h0 = relu(bn(x@W + b)) ----
  gemm_k128<8, 2, false><<<GB_GEMM, B256, 0, stream>>>(x, g_fc_w, h,
                                                       g_fc_b, g_bn0g, g_bn0b);
  for (int l = 0; l < 3; l++) {
    gemm_k128<8, 0, false><<<GB_GEMM, B256, 0, stream>>>(h, gw[l], hw,
                                                         nullptr, nullptr, nullptr);
    gcn_init_agg<<<GB_N32, B256, 0, stream>>>(hw, dis, agg);        // self loops
    gcn_edge<<<NEDGES / 8, B256, 0, stream>>>(ei, dis, hw, agg);    // L2-resident
    gcn_post<<<GB_N32, B256, 0, stream>>>(agg, gb[l], gbng[l], gbnb[l], h);
  }

  // ---- transformer branch (sorted layout) ----
  gemm_k128<8, 1, false><<<GB_GEMM, B256, 0, stream>>>(xs, t_fc_w, ts,
                                                       t_fc_b, nullptr, nullptr);
  ln_relu_kernel<<<NNODES / 8, B256, 0, stream>>>(ts, t_ln0g, t_ln0b);
  for (int l = 0; l < 2; l++) {
    long nz = NGRAPH * 128 * 128 + NGRAPH * 128 + 16;
    zero_f32<<<(int)((nz + 255) / 256), B256, 0, stream>>>(kvs, nz);
    gemm_k128<8, 0, false><<<GB_GEMM, B256, 0, stream>>>(ts, tq[l], qb,
                                                         nullptr, nullptr, nullptr);
    gemm_k128<8, 0, false><<<GB_GEMM, B256, 0, stream>>>(ts, tk[l], kb,
                                                         nullptr, nullptr, nullptr);
    gemm_k128<8, 0, false><<<GB_GEMM, B256, 0, stream>>>(ts, tv[l], vb,
                                                         nullptr, nullptr, nullptr);
    sumsq_kernel<<<1024, B256, 0, stream>>>(qb, &scal[0]);
    sumsq_kernel<<<1024, B256, 0, stream>>>(kb, &scal[1]);
    gemm_tn_kvs<<<NGRAPH * 8, B256, 0, stream>>>(kb, vb, kvs);
    colsum_kernel<<<NGRAPH * 8, dim3(128), 0, stream>>>(kb, ksum);
    gemm_k128<8, 0, true><<<GB_GEMM, B256, 0, stream>>>(qb, kvs, P,   // P aliases kb
                                                        nullptr, nullptr, nullptr);
    attn_combine<<<NNODES / 8, B256, 0, stream>>>(P, qb, vb, ts, ksum, scal,
                                                  tlng[l], tlnb[l]);
  }

  // ---- combine + classifier + log_softmax ----
  combine_out<<<GB_N32, B256, 0, stream>>>(h, ts, revb, outc);
  gemm_k128<4, 1, false><<<GB_GEMM, B256, 0, stream>>>(outc, fc_w, logits,
                                                       fc_b, nullptr, nullptr);
  log_softmax64<<<NNODES / 8, B256, 0, stream>>>(logits, (float*)d_out);
}
